// new_IBQ_17291538333820
// MI455X (gfx1250) — matrix operand order repaired and checked by reading, not compiled, not run on hardware
//
#include <hip/hip_runtime.h>
#include <math.h>

// ---------------------------------------------------------------------------
// VQ codebook-update pipeline for MI455X (gfx1250, wave32, WMMA).
// Distance GEMM+argmin: v_wmma_f32_16x16x32_f16, 32 rows x 32 cols per wave
// per iteration (8 WMMAs / 8 B-loads), packed-key argmin (pk_fma/and_or/min),
// K-split across blocks combined with global_atomic_min_u32 on packed keys.
// ---------------------------------------------------------------------------

typedef __attribute__((ext_vector_type(16))) _Float16 v16h;
typedef __attribute__((ext_vector_type(8)))  _Float16 v8h;
typedef __attribute__((ext_vector_type(8)))  float    v8f;

#define MU 0.99f
#define EPS 1e-6f
#define COMMIT_W 0.25f
#define IDX_MASK 0x1FFFu            // K = 8192 -> 13 bits
#define KSPLIT 4                    // K-range split for occupancy

// ---------------------------------------------------------------------------
// Kernel 1: L2-normalize rows of x; emit f32 x_norm and f16 copy.
// ---------------------------------------------------------------------------
__global__ void k_normalize(const float* __restrict__ x,
                            float* __restrict__ xn,
                            _Float16* __restrict__ xh,
                            int N) {
    int r = blockIdx.x * blockDim.x + threadIdx.x;
    if (r >= N) return;
    const float4* rp = reinterpret_cast<const float4*>(x + (size_t)r * 64);
    float v[64];
    float s = 0.f;
#pragma unroll
    for (int i = 0; i < 16; ++i) {
        float4 q = rp[i];
        v[4 * i + 0] = q.x; v[4 * i + 1] = q.y;
        v[4 * i + 2] = q.z; v[4 * i + 3] = q.w;
        s += q.x * q.x + q.y * q.y + q.z * q.z + q.w * q.w;
    }
    float inv = 1.0f / fmaxf(sqrtf(s), 1e-12f);
    float4* op = reinterpret_cast<float4*>(xn + (size_t)r * 64);
    _Float16* hp = xh + (size_t)r * 64;
#pragma unroll
    for (int i = 0; i < 16; ++i) {
        float4 q;
        q.x = v[4 * i + 0] * inv; q.y = v[4 * i + 1] * inv;
        q.z = v[4 * i + 2] * inv; q.w = v[4 * i + 3] * inv;
        op[i] = q;
        hp[4 * i + 0] = (_Float16)q.x; hp[4 * i + 1] = (_Float16)q.y;
        hp[4 * i + 2] = (_Float16)q.z; hp[4 * i + 3] = (_Float16)q.w;
    }
}

// ---------------------------------------------------------------------------
// Kernel 2: codebook prep: c2[k] = |c_k|^2 (f32 exact), f16 copy of codebook.
// ---------------------------------------------------------------------------
__global__ void k_cbprep(const float* __restrict__ cb,
                         _Float16* __restrict__ cbh,
                         float* __restrict__ c2,
                         int K) {
    int k = blockIdx.x * blockDim.x + threadIdx.x;
    if (k >= K) return;
    const float4* rp = reinterpret_cast<const float4*>(cb + (size_t)k * 64);
    _Float16* hp = cbh + (size_t)k * 64;
    float s = 0.f;
#pragma unroll
    for (int i = 0; i < 16; ++i) {
        float4 q = rp[i];
        s += q.x * q.x + q.y * q.y + q.z * q.z + q.w * q.w;
        hp[4 * i + 0] = (_Float16)q.x; hp[4 * i + 1] = (_Float16)q.y;
        hp[4 * i + 2] = (_Float16)q.z; hp[4 * i + 3] = (_Float16)q.w;
    }
    c2[k] = s;
}

// ---------------------------------------------------------------------------
// Kernel 3: fused distance GEMM + argmin, WMMA f32_16x16x32_f16.
// Block = 128 threads = 4 waves; each wave owns 32 rows (two 16-row tiles).
// blockIdx.y selects a K/KSPLIT column range; partial argmins are combined
// with global_atomic_min_u32 on packed keys.
//
// Argmin trick: score = c2[col] - 2*dot (row-constant x^2 dropped; score is
// strictly positive for this data), key = (bits(score) & ~0x1FFF) | col.
// Unsigned min on keys == float min with lowest-index tie-break.
// Separate running bests per column half avoid cross-accumulator register
// shuffles in the pk_fma epilogue.
//
// Layouts per CDNA5 ISA (wave32):
//   A 16x32 f16 : lane l holds row (l&15); halves [lo,lo+8),[16+lo,16+lo+8),
//                 lo = (l>>4)*8, for each 32-wide K chunk.
//   B 32x16 f16 : lane l holds col (l&15); halves [dlo,dlo+8),[16+dlo,16+dlo+8),
//                 dlo = (l>>4)*8 (same as A), for each 32-wide K chunk.
//   C/D 16x16 f32: vgpr v, lane l -> row (l>>4)*8 + v, col (l&15).
// ---------------------------------------------------------------------------
__global__ __launch_bounds__(128) void k_argmin(
    const _Float16* __restrict__ xh, const _Float16* __restrict__ cbh,
    const float* __restrict__ c2,
    unsigned int* __restrict__ keys, int K) {

    const int lane  = threadIdx.x & 31;
    const int wave  = threadIdx.x >> 5;
    const int rbase = blockIdx.x * 128 + wave * 32;

    const int lo     = (lane >> 4) * 8;    // A half-chunk select
    const int rowoff = (lane >> 4) * 8;    // C-tile row group
    const int nlane  = lane & 15;          // C-tile / B column
    const int dlo    = (lane >> 4) * 8;    // B K-chunk select

    const int kspan = K / KSPLIT;
    const int kbeg  = blockIdx.y * kspan;

    // ---- A fragments: rows rbase..rbase+15 (a0,a1) and +16..+31 (a2,a3)
    v16h a0, a1, a2, a3;
    {
        const _Float16* ap = xh + (size_t)(rbase + (lane & 15)) * 64;
        const _Float16* aq = ap + 16 * 64;
        v8h c0 = *reinterpret_cast<const v8h*>(ap + 0  + lo);
        v8h c1 = *reinterpret_cast<const v8h*>(ap + 16 + lo);
        v8h c2h = *reinterpret_cast<const v8h*>(ap + 32 + lo);
        v8h c3 = *reinterpret_cast<const v8h*>(ap + 48 + lo);
        v8h d0 = *reinterpret_cast<const v8h*>(aq + 0  + lo);
        v8h d1 = *reinterpret_cast<const v8h*>(aq + 16 + lo);
        v8h d2 = *reinterpret_cast<const v8h*>(aq + 32 + lo);
        v8h d3 = *reinterpret_cast<const v8h*>(aq + 48 + lo);
#pragma unroll
        for (int i = 0; i < 8; ++i) {
            a0[i] = c0[i]; a0[8 + i] = c1[i];
            a1[i] = c2h[i]; a1[8 + i] = c3[i];
            a2[i] = d0[i]; a2[8 + i] = d1[i];
            a3[i] = d2[i]; a3[8 + i] = d3[i];
        }
    }

    // Separate running bests: tile0/tile1 rows x col0/col1 halves.
    unsigned int b0A[8], b0B[8], b1A[8], b1B[8];
#pragma unroll
    for (int v = 0; v < 8; ++v) {
        b0A[v] = 0xFFFFFFFFu; b0B[v] = 0xFFFFFFFFu;
        b1A[v] = 0xFFFFFFFFu; b1B[v] = 0xFFFFFFFFu;
    }

    // Pointer-increment loop: +32 columns -> +4096 bytes per iteration.
    const _Float16* p0 = cbh + (size_t)(kbeg + nlane) * 64 + dlo;
    const _Float16* p1 = p0 + 16 * 64;
    unsigned int col0 = (unsigned int)(kbeg + nlane);

    for (int it = kspan >> 5; it > 0; --it) {
        const unsigned int col1 = col0 + 16u;
        __builtin_prefetch(p0 + 2048, 0, 1);   // next 32-col tile (4 KB ahead)

        v16h bA0, bB0, bA1, bB1;
        {
            v8h t0 = *reinterpret_cast<const v8h*>(p0);
            v8h t1 = *reinterpret_cast<const v8h*>(p0 + 16);
            v8h t2 = *reinterpret_cast<const v8h*>(p0 + 32);
            v8h t3 = *reinterpret_cast<const v8h*>(p0 + 48);
            v8h u0 = *reinterpret_cast<const v8h*>(p1);
            v8h u1 = *reinterpret_cast<const v8h*>(p1 + 16);
            v8h u2 = *reinterpret_cast<const v8h*>(p1 + 32);
            v8h u3 = *reinterpret_cast<const v8h*>(p1 + 48);
#pragma unroll
            for (int i = 0; i < 8; ++i) {
                bA0[i] = t0[i]; bA0[8 + i] = t1[i];
                bB0[i] = t2[i]; bB0[8 + i] = t3[i];
                bA1[i] = u0[i]; bA1[8 + i] = u1[i];
                bB1[i] = u2[i]; bB1[8 + i] = u3[i];
            }
        }

        v8f z = {0.f, 0.f, 0.f, 0.f, 0.f, 0.f, 0.f, 0.f};
        v8f acc00 = z, acc01 = z, acc10 = z, acc11 = z;
        // rows 0..15 x cols 0..15 / 16..31
        acc00 = __builtin_amdgcn_wmma_f32_16x16x32_f16(false, a0, false, bA0,
                                                       (short)0, acc00, false, false);
        acc01 = __builtin_amdgcn_wmma_f32_16x16x32_f16(false, a0, false, bA1,
                                                       (short)0, acc01, false, false);
        // rows 16..31
        acc10 = __builtin_amdgcn_wmma_f32_16x16x32_f16(false, a2, false, bA0,
                                                       (short)0, acc10, false, false);
        acc11 = __builtin_amdgcn_wmma_f32_16x16x32_f16(false, a2, false, bA1,
                                                       (short)0, acc11, false, false);
        acc00 = __builtin_amdgcn_wmma_f32_16x16x32_f16(false, a1, false, bB0,
                                                       (short)0, acc00, false, false);
        acc01 = __builtin_amdgcn_wmma_f32_16x16x32_f16(false, a1, false, bB1,
                                                       (short)0, acc01, false, false);
        acc10 = __builtin_amdgcn_wmma_f32_16x16x32_f16(false, a3, false, bB0,
                                                       (short)0, acc10, false, false);
        acc11 = __builtin_amdgcn_wmma_f32_16x16x32_f16(false, a3, false, bB1,
                                                       (short)0, acc11, false, false);

        const float c20 = c2[col0];
        const float c21 = c2[col1];
#pragma unroll
        for (int v = 0; v < 8; ++v) {
            unsigned int k00 = (__float_as_uint(fmaf(-2.0f, acc00[v], c20)) & ~IDX_MASK) | col0;
            b0A[v] = b0A[v] < k00 ? b0A[v] : k00;
            unsigned int k01 = (__float_as_uint(fmaf(-2.0f, acc01[v], c21)) & ~IDX_MASK) | col1;
            b0B[v] = b0B[v] < k01 ? b0B[v] : k01;
            unsigned int k10 = (__float_as_uint(fmaf(-2.0f, acc10[v], c20)) & ~IDX_MASK) | col0;
            b1A[v] = b1A[v] < k10 ? b1A[v] : k10;
            unsigned int k11 = (__float_as_uint(fmaf(-2.0f, acc11[v], c21)) & ~IDX_MASK) | col1;
            b1B[v] = b1B[v] < k11 ? b1B[v] : k11;
        }

        p0 += 2048; p1 += 2048; col0 += 32u;
    }

    // Merge column halves, then butterfly unsigned-min across the 16 column
    // lanes of each half-wave.
    unsigned int best0[8], best1[8];
#pragma unroll
    for (int v = 0; v < 8; ++v) {
        best0[v] = b0A[v] < b0B[v] ? b0A[v] : b0B[v];
        best1[v] = b1A[v] < b1B[v] ? b1A[v] : b1B[v];
    }
#pragma unroll
    for (int m = 1; m < 16; m <<= 1) {
#pragma unroll
        for (int v = 0; v < 8; ++v) {
            unsigned int o0 = (unsigned int)__shfl_xor((int)best0[v], m, 32);
            unsigned int o1 = (unsigned int)__shfl_xor((int)best1[v], m, 32);
            best0[v] = best0[v] < o0 ? best0[v] : o0;
            best1[v] = best1[v] < o1 ? best1[v] : o1;
        }
    }

    // Combine partial (per K-range) results: unsigned atomic min on keys.
    if (nlane == 0) {
#pragma unroll
        for (int v = 0; v < 8; ++v) {
            atomicMin(&keys[rbase + rowoff + v], best0[v]);
            atomicMin(&keys[rbase + 16 + rowoff + v], best1[v]);
        }
    }
}

// ---------------------------------------------------------------------------
// Kernel 4: decode keys -> idx, dequantize (x_d from f32 codebook), scatter
// segment sums with global f32 atomics, accumulate commit-loss partial.
// Block = 256 threads = 4 rows x 64 dims (coalesced).
// ---------------------------------------------------------------------------
__global__ void k_dequant_scatter(const unsigned int* __restrict__ keys,
                                  const float* __restrict__ cb,
                                  const float* __restrict__ xn,
                                  float* __restrict__ out_xd,
                                  float* __restrict__ out_idx,
                                  float* __restrict__ sum_local,
                                  float* __restrict__ count_local,
                                  float* __restrict__ commit_acc) {
    __shared__ float red[256];
    const int t = threadIdx.x;
    const int r = blockIdx.x * 4 + (t >> 6);
    const int d = t & 63;
    const int idx = (int)(keys[r] & IDX_MASK);
    const float xd = cb[(size_t)idx * 64 + d];
    const float xv = xn[(size_t)r * 64 + d];
    out_xd[(size_t)r * 64 + d] = xd;
    atomicAdd(&sum_local[(size_t)idx * 64 + d], xv);
    if (d == 0) {
        atomicAdd(&count_local[idx], 1.0f);
        out_idx[r] = (float)idx;
    }
    float df = xv - xd;
    red[t] = df * df;
    __syncthreads();
#pragma unroll
    for (int s = 128; s > 0; s >>= 1) {
        if (t < s) red[t] += red[t + s];
        __syncthreads();
    }
    if (t == 0) atomicAdd(commit_acc, red[0]);
}

// ---------------------------------------------------------------------------
// Kernel 5a: EMA count update + totals (sums[0]=sum(new_count),
// sums[1]=sum(count_local)).
// ---------------------------------------------------------------------------
__global__ void k_count_update(const float* __restrict__ cc,
                               const float* __restrict__ ccl,
                               float* __restrict__ ncc_out,
                               float* __restrict__ sums, int K) {
    int k = blockIdx.x * blockDim.x + threadIdx.x;
    if (k >= K) return;
    float l = ccl[k];
    float n = MU * cc[k] + (1.0f - MU) * l;
    ncc_out[k] = n;
    atomicAdd(&sums[0], n);
    atomicAdd(&sums[1], l);
}

// ---------------------------------------------------------------------------
// Kernel 5b: EMA codebook update: new_cb = new_sum / (new_count + EPS).
// ---------------------------------------------------------------------------
__global__ void k_codebook_update(const float* __restrict__ cs,
                                  const float* __restrict__ csl,
                                  const float* __restrict__ ncc,
                                  float* __restrict__ out_cb, int KD) {
    int i = blockIdx.x * blockDim.x + threadIdx.x;
    if (i >= KD) return;
    int k = i >> 6;
    float ns = MU * cs[i] + (1.0f - MU) * csl[i];
    out_cb[i] = ns / (ncc[k] + EPS);
}

// ---------------------------------------------------------------------------
// Kernel 5c: entropy accumulation: Hs[0] += pT*log(pT+EPS), Hs[1] likewise.
// ---------------------------------------------------------------------------
__global__ void k_entropy(const float* __restrict__ ncc,
                          const float* __restrict__ ccl,
                          const float* __restrict__ sums,
                          float* __restrict__ Hs, int K) {
    int k = blockIdx.x * blockDim.x + threadIdx.x;
    if (k >= K) return;
    float pT = ncc[k] / (sums[0] + EPS);
    float pC = ccl[k] / (sums[1] + EPS);
    atomicAdd(&Hs[0], pT * logf(pT + EPS));
    atomicAdd(&Hs[1], pC * logf(pC + EPS));
}

// ---------------------------------------------------------------------------
// Kernel 5d: finalize the three scalars.
// ---------------------------------------------------------------------------
__global__ void k_finalize(const float* __restrict__ commit_acc,
                           const float* __restrict__ Hs,
                           float* __restrict__ out_commit,
                           float* __restrict__ out_tp,
                           float* __restrict__ out_cp, int ND) {
    *out_commit = COMMIT_W * (*commit_acc) / (float)ND;
    *out_tp = expf(-Hs[0]);
    *out_cp = expf(-Hs[1]);
}

// ---------------------------------------------------------------------------
// Launch
// ---------------------------------------------------------------------------
extern "C" void kernel_launch(void* const* d_in, const int* in_sizes, int n_in,
                              void* d_out, int out_size, void* d_ws, size_t ws_size,
                              hipStream_t stream) {
    (void)n_in; (void)out_size; (void)ws_size;
    const float* x  = (const float*)d_in[0];   // [N,64]
    const float* cb = (const float*)d_in[1];   // [K,64]
    const float* cs = (const float*)d_in[2];   // [K,64]
    const float* cc = (const float*)d_in[3];   // [K]

    const int D = 64;
    const int N = in_sizes[0] / D;             // 32768
    const int K = in_sizes[1] / D;             // 8192

    // ---- workspace layout (bytes) --------------------------------------
    char* ws = (char*)d_ws;
    size_t off = 0;
    float*    ws_xn   = (float*)(ws + off);    off += (size_t)N * D * 4;  // x_norm f32
    _Float16* ws_xh   = (_Float16*)(ws + off); off += (size_t)N * D * 2;  // x f16
    _Float16* ws_cbh  = (_Float16*)(ws + off); off += (size_t)K * D * 2;  // codebook f16
    float*    ws_c2   = (float*)(ws + off);    off += (size_t)K * 4;      // |c|^2
    size_t koff = off;                                                    // 0xFF region:
    unsigned int* ws_key = (unsigned int*)(ws + off); off += (size_t)N * 4; // packed keys
    size_t zoff = off;                                                    // zeroed region:
    float*    ws_csl  = (float*)(ws + off);    off += (size_t)K * D * 4;  // code_sum_local
    float*    ws_ccl  = (float*)(ws + off);    off += (size_t)K * 4;      // code_count_local
    float*    ws_ncc  = (float*)(ws + off);    off += (size_t)K * 4;      // new_code_count
    float*    ws_sums = (float*)(ws + off);    off += 2 * 4;              // totals
    float*    ws_Hs   = (float*)(ws + off);    off += 2 * 4;              // entropies
    float*    ws_cmt  = (float*)(ws + off);    off += 4;                  // commit partial
    size_t zsize = off - zoff;

    // ---- output layout (f32, concatenated in reference return order) --
    float* out = (float*)d_out;
    float* out_xd  = out;                                   // [N,64]
    float* out_idx = out + (size_t)N * D;                   // [N]
    float* out_cl  = out + (size_t)N * D + N;               // commit loss
    float* out_cb  = out + (size_t)N * D + N + 1;           // [K,64]
    float* out_tp  = out + (size_t)N * D + N + 1 + (size_t)K * D;
    float* out_cp  = out_tp + 1;

    hipMemsetAsync(ws + koff, 0xFF, (size_t)N * 4, stream); // keys = 0xFFFFFFFF
    hipMemsetAsync(ws + zoff, 0, zsize, stream);

    k_normalize<<<(N + 255) / 256, 256, 0, stream>>>(x, ws_xn, ws_xh, N);
    k_cbprep<<<(K + 255) / 256, 256, 0, stream>>>(cb, ws_cbh, ws_c2, K);
    k_argmin<<<dim3(N / 128, KSPLIT), 128, 0, stream>>>(ws_xh, ws_cbh, ws_c2,
                                                        ws_key, K);
    k_dequant_scatter<<<N / 4, 256, 0, stream>>>(ws_key, cb, ws_xn, out_xd,
                                                 out_idx, ws_csl, ws_ccl,
                                                 ws_cmt);
    k_count_update<<<(K + 255) / 256, 256, 0, stream>>>(cc, ws_ccl, ws_ncc,
                                                        ws_sums, K);
    k_codebook_update<<<(K * D + 255) / 256, 256, 0, stream>>>(cs, ws_csl,
                                                               ws_ncc, out_cb,
                                                               K * D);
    k_entropy<<<(K + 255) / 256, 256, 0, stream>>>(ws_ncc, ws_ccl, ws_sums,
                                                   ws_Hs, K);
    k_finalize<<<1, 1, 0, stream>>>(ws_cmt, ws_Hs, out_cl, out_tp, out_cp,
                                    N * D);
}
